// WindowAttention_31920196944320
// MI455X (gfx1250) — matrix operand order repaired and checked by reading, not compiled, not run on hardware
//
#include <hip/hip_runtime.h>

typedef __attribute__((ext_vector_type(16))) _Float16 v16h;
typedef __attribute__((ext_vector_type(8)))  _Float16 v8h;
typedef __attribute__((ext_vector_type(4)))  _Float16 v4h;
typedef __attribute__((ext_vector_type(8)))  float    v8f;

#define NWIN 4096
#define NTOK 49
#define NPAD 64
#define CDIM 256
#define NHEAD 8
#define HDIM 32
#define ROW  264   // row stride (halfs) for xs/qs/ks/ao: 256 + 8 pad (16B) -> no bank conflicts
#define VROW 72    // row stride (halfs) for V^T: 64 + 8 pad

// Prevent LICM from hoisting LDS tile loads across outer tile loops
// (the hoisted form needs ~256 extra VGPRs and spills to scratch).
#define TILE_LOOP_FENCE() asm volatile("" ::: "memory")

// ---------------- WMMA helpers (CDNA5 gfx1250, wave32) ----------------

__device__ __forceinline__ v8f wmma_f16(v16h a, v16h b, v8f c) {
  // 8 args: (neg_a, A, neg_b, B, c_mod, C, reuse_a, reuse_b)
  return __builtin_amdgcn_wmma_f32_16x16x32_f16(false, a, false, b, (short)0, c, false, false);
}

// A-matrix 16x32 f16 tile load. Per ISA layout: lane holds row m=lane%16,
// K halfs 0..7 at k0=(lane/16)*8 and halfs 8..15 at k0+16.
// `p` points at element [m][k0]; second chunk is 16 halfs further.
__device__ __forceinline__ v16h ld_a16(const _Float16* p) {
  v8h lo = *(const v8h*)p;
  v8h hi = *(const v8h*)(p + 16);
  return __builtin_shufflevector(lo, hi, 0,1,2,3,4,5,6,7,8,9,10,11,12,13,14,15);
}

// B-matrix 32x16 f16 tile load. Lane holds column n=lane%16,
// K halfs 0..7 at k0=(lane/16)*8 and halfs 8..15 at k0+16. `p` points at element [n][k0].
__device__ __forceinline__ v16h ld_b16(const _Float16* p) {
  v8h lo = *(const v8h*)p;
  v8h hi = *(const v8h*)(p + 16);
  return __builtin_shufflevector(lo, hi, 0,1,2,3,4,5,6,7,8,9,10,11,12,13,14,15);
}

__device__ __forceinline__ v8f vzero8() {
  v8f z;
#pragma unroll
  for (int e = 0; e < 8; ++e) z[e] = 0.0f;
  return z;
}

// ---------------- prologue: f32->f16 weights + gathered padded bias ----------------
// biasP[h][i][j] (i,j in [0,64)): rel-pos bias for valid (i<49,j<49),
// -1e30 for j>=49 (kills padded keys in softmax), 0 otherwise.

__global__ void prep_kernel(const float* __restrict__ wqkv, const float* __restrict__ wproj,
                            const float* __restrict__ btab, const int* __restrict__ ridx,
                            _Float16* __restrict__ wq_h, _Float16* __restrict__ wp_h,
                            float* __restrict__ biasP) {
  int t = blockIdx.x * 256 + threadIdx.x;
  if (t < 196608) {                       // 768*256
    wq_h[t] = (_Float16)wqkv[t];
  } else if (t < 262144) {                // + 256*256
    int i = t - 196608;
    wp_h[i] = (_Float16)wproj[i];
  } else if (t < 294912) {                // + 8*64*64
    int i = t - 262144;
    int h = i >> 12, r = i & 4095, ii = r >> 6, jj = r & 63;
    float v;
    if (ii < 49 && jj < 49) v = btab[ridx[ii * 49 + jj] * NHEAD + h];
    else                    v = (jj >= 49) ? -1.0e30f : 0.0f;
    biasP[i] = v;
  }
}

// ---------------- fused window attention: one block per window ----------------
// launch_bounds(256, 4): LDS (154KB) allows 2 workgroups per 320KB WGP = 4 waves/SIMD.
// With the per-iteration fences the kernel needs ~120 VGPRs, so this fits without spills.

__global__ void __launch_bounds__(256, 4)
win_attn_kernel(const float* __restrict__ x, const float* __restrict__ mask,
                const float* __restrict__ bqkv, const float* __restrict__ bproj,
                const _Float16* __restrict__ wq, const _Float16* __restrict__ wp,
                const float* __restrict__ biasP, float* __restrict__ out) {
  __shared__ __align__(16) _Float16 xs[NPAD * ROW];   // x staged f16; reused as attn-out
  __shared__ __align__(16) _Float16 qs[NPAD * ROW];   // Q row-major [token][feat]
  __shared__ __align__(16) _Float16 ks[NPAD * ROW];   // K row-major [token][feat]
  __shared__ __align__(16) _Float16 vt[CDIM * VROW];  // V^T [feat][token]
  __shared__ __align__(16) float    smask[NPAD * NPAD];

  const int w    = blockIdx.x;
  const int tid  = threadIdx.x;
  const int lane = tid & 31;
  const int wv   = tid >> 5;       // wave id 0..7
  const int lr   = lane & 15;      // lane row/col within tile
  const int hl   = lane >> 4;      // half-wave select

  // ---- stage x (f32 -> f16) into LDS, zero pad rows 49..63 ----
  {
    const float4* xp = (const float4*)(x + (size_t)w * (NTOK * CDIM));
    for (int id = tid; id < 3136; id += 256) {      // 49*256/4
      int i = id >> 6, c4 = id & 63;
      float4 v = xp[id];
      v4h h4; h4[0] = (_Float16)v.x; h4[1] = (_Float16)v.y;
              h4[2] = (_Float16)v.z; h4[3] = (_Float16)v.w;
      *(v4h*)&xs[i * ROW + c4 * 4] = h4;
    }
    for (int id = tid; id < 960; id += 256) {       // 15 pad rows * 64 chunks
      int i = 49 + (id >> 6), c4 = id & 63;
      v4h h4; h4[0] = h4[1] = h4[2] = h4[3] = (_Float16)0.0f;
      *(v4h*)&xs[i * ROW + c4 * 4] = h4;
    }
    // padded border of the mask tile (address-disjoint from the async DMA below)
    for (int id = tid; id < 1695; id += 256) {
      int i, j;
      if (id < 735) { i = id / 15; j = 49 + (id - i * 15); }          // rows 0..48, cols 49..63
      else { int r = id - 735; i = 49 + (r >> 6); j = r & 63; }       // rows 49..63
      smask[i * NPAD + j] = 0.0f;
    }
    // valid 49x49 mask region: async global->LDS DMA (ASYNCcnt), no VGPR round-trip
    const float* mp = mask + (size_t)w * (NTOK * NTOK);
    for (int id = tid; id < NTOK * NTOK; id += 256) {
      int i = id / NTOK, j = id - i * NTOK;
      unsigned int      lds = (unsigned int)(uintptr_t)&smask[i * NPAD + j];
      unsigned long long ga = (unsigned long long)(uintptr_t)(mp + id);
      asm volatile("global_load_async_to_lds_b32 %0, %1, off"
                   :: "v"(lds), "v"(ga) : "memory");
    }
    asm volatile("s_wait_asynccnt 0x0" ::: "memory");
  }
  __syncthreads();

  // ---- QKV projection: [64 x 256] x [256 x 768]^T, each wave owns 6 otiles ----
  for (int ot = wv; ot < 48; ot += 8) {
    TILE_LOOP_FENCE();
    v8f acc[4];
#pragma unroll
    for (int it = 0; it < 4; ++it) acc[it] = vzero8();
    const _Float16* bbase = wq + (ot * 16 + lr) * CDIM;
#pragma unroll
    for (int kk = 0; kk < 8; ++kk) {
      v16h B = ld_b16(bbase + kk * 32 + hl * 8);
#pragma unroll
      for (int it = 0; it < 4; ++it) {
        v16h A = ld_a16(&xs[(it * 16 + lr) * ROW + kk * 32 + hl * 8]);
        acc[it] = wmma_f16(A, B, acc[it]);
      }
    }
    float bo = bqkv[ot * 16 + lr];                 // D: lane holds fixed output column
    if (ot < 32) {
      _Float16* dst = (ot < 16) ? qs : ks;
      int f = (ot & 15) * 16 + lr;
#pragma unroll
      for (int it = 0; it < 4; ++it)
#pragma unroll
        for (int e = 0; e < 8; ++e)
          dst[(it * 16 + hl * 8 + e) * ROW + f] = (_Float16)(acc[it][e] + bo);
    } else {
      int f = (ot - 32) * 16 + lr;                 // store V transposed: contiguous b128
#pragma unroll
      for (int it = 0; it < 4; ++it) {
        v8h pk;
#pragma unroll
        for (int e = 0; e < 8; ++e) pk[e] = (_Float16)(acc[it][e] + bo);
        *(v8h*)&vt[f * VROW + it * 16 + hl * 8] = pk;
      }
    }
  }
  __syncthreads();

  // ---- attention: wave wv owns head h = wv ----
  {
    const int h = wv;
    const float scale = 0.17677669529663687f;      // 1/sqrt(32)
    const float* bph = biasP + h * (NPAD * NPAD);
    _Float16* ao = xs;                             // xs dead -> reuse as attn-out
#pragma unroll 1
    for (int it = 0; it < 4; ++it) {
      TILE_LOOP_FENCE();
      const int i = it * 16 + lr;                  // query token, fixed per lane
      v16h Bq = ld_b16(&qs[i * ROW + h * HDIM + hl * 8]);
      v8f st[4];
#pragma unroll
      for (int jt = 0; jt < 4; ++jt) {             // S^T = K * Q^T -> lane holds row i of S
        v16h A = ld_a16(&ks[(jt * 16 + lr) * ROW + h * HDIM + hl * 8]);
        st[jt] = wmma_f16(A, Bq, vzero8());
      }
      float vr[4][8];
      float mx = -3.0e38f;
#pragma unroll
      for (int jt = 0; jt < 4; ++jt) {             // j = jt*16 + hl*8 + e
        const float4* bp = (const float4*)(bph + i * NPAD + jt * 16 + hl * 8);
        const float4* sp = (const float4*)(&smask[i * NPAD + jt * 16 + hl * 8]);
        float4 b0 = bp[0], b1 = bp[1], m0 = sp[0], m1 = sp[1];
        float bb[8] = { b0.x + m0.x, b0.y + m0.y, b0.z + m0.z, b0.w + m0.w,
                        b1.x + m1.x, b1.y + m1.y, b1.z + m1.z, b1.w + m1.w };
#pragma unroll
        for (int e = 0; e < 8; ++e) {
          float v = st[jt][e] * scale + bb[e];
          vr[jt][e] = v;
          mx = fmaxf(mx, v);
        }
      }
      mx = fmaxf(mx, __shfl_xor(mx, 16, 32));      // other half of row i lives in lane^16
      float sum = 0.0f;
#pragma unroll
      for (int jt = 0; jt < 4; ++jt)
#pragma unroll
        for (int e = 0; e < 8; ++e) {
          float p = __expf(vr[jt][e] - mx);
          vr[jt][e] = p;
          sum += p;
        }
      sum += __shfl_xor(sum, 16, 32);
      float inv = 1.0f / sum;
      // P registers map directly onto A-matrix halfs: jt0->0..7, jt1->8..15 (tile j<32), etc.
      v16h A0, A1;
#pragma unroll
      for (int e = 0; e < 8; ++e) {
        A0[e]     = (_Float16)(vr[0][e] * inv);
        A0[8 + e] = (_Float16)(vr[1][e] * inv);
        A1[e]     = (_Float16)(vr[2][e] * inv);
        A1[8 + e] = (_Float16)(vr[3][e] * inv);
      }
      v8f oacc[2] = { vzero8(), vzero8() };
#pragma unroll
      for (int kt = 0; kt < 2; ++kt) {
        v16h Ap = kt ? A1 : A0;
#pragma unroll
        for (int dt = 0; dt < 2; ++dt) {
          v16h Bv = ld_b16(&vt[(h * HDIM + dt * 16 + lr) * VROW + kt * 32 + hl * 8]);
          oacc[dt] = wmma_f16(Ap, Bv, oacc[dt]);
        }
      }
#pragma unroll
      for (int dt = 0; dt < 2; ++dt)
#pragma unroll
        for (int e = 0; e < 8; ++e)
          ao[(it * 16 + hl * 8 + e) * ROW + h * HDIM + dt * 16 + lr] = (_Float16)oacc[dt][e];
    }
  }
  __syncthreads();

  // ---- output projection: [64 x 256] x [256 x 256]^T, 2 otiles per wave ----
  {
    const _Float16* ao = xs;
    float* ob = out + (size_t)w * (NTOK * CDIM);
#pragma unroll 1
    for (int ot = wv * 2; ot < wv * 2 + 2; ++ot) {
      TILE_LOOP_FENCE();
      v8f acc[4];
#pragma unroll
      for (int it = 0; it < 4; ++it) acc[it] = vzero8();
      const _Float16* bbase = wp + (ot * 16 + lr) * CDIM;
#pragma unroll
      for (int kk = 0; kk < 8; ++kk) {
        v16h B = ld_b16(bbase + kk * 32 + hl * 8);
#pragma unroll
        for (int it = 0; it < 4; ++it) {
          v16h A = ld_a16(&ao[(it * 16 + lr) * ROW + kk * 32 + hl * 8]);
          acc[it] = wmma_f16(A, B, acc[it]);
        }
      }
      float bo = bproj[ot * 16 + lr];
      int o = ot * 16 + lr;
#pragma unroll
      for (int it = 0; it < 4; ++it)
#pragma unroll
        for (int e = 0; e < 8; ++e) {
          int i = it * 16 + hl * 8 + e;
          if (i < NTOK) ob[i * CDIM + o] = acc[it][e] + bo;
        }
    }
  }
}

extern "C" void kernel_launch(void* const* d_in, const int* in_sizes, int n_in,
                              void* d_out, int out_size, void* d_ws, size_t ws_size,
                              hipStream_t stream) {
  const float* x     = (const float*)d_in[0];
  const float* mask  = (const float*)d_in[1];
  const float* wqkv  = (const float*)d_in[2];
  const float* bqkv  = (const float*)d_in[3];
  const float* wproj = (const float*)d_in[4];
  const float* bproj = (const float*)d_in[5];
  const float* btab  = (const float*)d_in[6];
  const int*   ridx  = (const int*)d_in[7];

  _Float16* wq_h  = (_Float16*)d_ws;                          // 768*256*2 = 393216 B
  _Float16* wp_h  = (_Float16*)((char*)d_ws + 393216);        // 256*256*2 = 131072 B
  float*    biasP = (float*)((char*)d_ws + 524288);           // 8*64*64*4 = 131072 B
  float*    outp  = (float*)d_out;

  prep_kernel<<<1152, 256, 0, stream>>>(wqkv, wproj, btab, ridx, wq_h, wp_h, biasP);
  win_attn_kernel<<<NWIN, 256, 0, stream>>>(x, mask, bqkv, bproj, wq_h, wp_h, biasP, outp);
}
